// PIGNN_29669634081206
// MI455X (gfx1250) — matrix operand order repaired and checked by reading, not compiled, not run on hardware
//
#include <hip/hip_runtime.h>

#define NN 50000
#define NE 200000
#define HDIM 128
#define NLAYERS 6
#define MSG_T 8          // edge tiles per block (msg kernel)
#define UPD_T 8          // node tiles per block (upd kernel)

typedef __attribute__((ext_vector_type(16))) __bf16 v16bf;
typedef __attribute__((ext_vector_type(8)))  float  v8f;

// ---------- helpers ----------

__device__ __forceinline__ unsigned short f2bf(float f) {
  unsigned u = __float_as_uint(f);
  u += 0x7FFFu + ((u >> 16) & 1u);           // round-to-nearest-even
  return (unsigned short)(u >> 16);
}

__device__ __forceinline__ uint2 f4_to_bf4(float4 f) {
  uint2 r;
  r.x = ((unsigned)f2bf(f.y) << 16) | (unsigned)f2bf(f.x);
  r.y = ((unsigned)f2bf(f.w) << 16) | (unsigned)f2bf(f.z);
  return r;
}

// A fragment (16x32 bf16, wave32) from an LDS tile staged as ushort rows.
// ISA layout: lane L: M=L&15, half=L>>4; VGPR p holds K = 2p + (p>=4?8:0) + 8*half (pairs).
__device__ __forceinline__ v16bf lds_a_frag(const unsigned short* lds, int stride_h,
                                            int kt, int lane) {
  const int m = lane & 15, half = lane >> 4;
  union { v16bf v; unsigned u[8]; } A;
  const unsigned* base = (const unsigned*)(lds + m * stride_h);   // stride_h even
  const int kb = (kt << 4) + (half << 2);                          // u32 units
#pragma unroll
  for (int p = 0; p < 8; ++p)
    A.u[p] = base[kb + p + ((p >= 4) ? 4 : 0)];
  return A.v;
}

// B fragment (32x16 bf16) from pre-packed weights: contiguous 16 bf16 per lane.
__device__ __forceinline__ v16bf ld_b_frag(const unsigned short* pk, int kt, int nt,
                                           int ntiles, int lane) {
  const unsigned short* p = pk + ((size_t)(((kt * ntiles + nt) << 5) + lane) << 4);
  union { v16bf v; uint4 q[2]; } B;
  B.q[0] = *(const uint4*)p;
  B.q[1] = *(const uint4*)(p + 8);
  return B.v;
}

__device__ __forceinline__ v8f wmma_bf16(v16bf a, v16bf b, v8f c) {
  return __builtin_amdgcn_wmma_f32_16x16x32_bf16(false, a, false, b, (short)0, c,
                                                 false, false);
}

// Pipelined GEMM: A from LDS, B resident in registers. Double-buffers the A frag
// so the kt+1 DS loads are in flight while WMMA kt executes.
template <int KT>
__device__ __forceinline__ v8f gemm_lds_regw(const unsigned short* Alds, int astride,
                                             const v16bf* w, int lane) {
  v8f acc = {0, 0, 0, 0, 0, 0, 0, 0};
  v16bf a = lds_a_frag(Alds, astride, 0, lane);
#pragma unroll
  for (int kt = 0; kt < KT - 1; ++kt) {
    v16bf an = lds_a_frag(Alds, astride, kt + 1, lane);
    acc = wmma_bf16(a, w[kt], acc);
    a = an;
  }
  return wmma_bf16(a, w[KT - 1], acc);
}

// Pipelined GEMM: A from LDS, B streamed from packed global weights (L2-resident).
template <int KT>
__device__ __forceinline__ v8f gemm_lds_gmemw(const unsigned short* Alds, int astride,
                                              const unsigned short* Wp, int ntiles,
                                              int wave, int lane) {
  v8f acc = {0, 0, 0, 0, 0, 0, 0, 0};
  v16bf a = lds_a_frag(Alds, astride, 0, lane);
  v16bf b = ld_b_frag(Wp, 0, wave, ntiles, lane);
#pragma unroll
  for (int kt = 0; kt < KT - 1; ++kt) {
    v16bf an = lds_a_frag(Alds, astride, kt + 1, lane);
    v16bf bn = ld_b_frag(Wp, kt + 1, wave, ntiles, lane);
    acc = wmma_bf16(a, b, acc);
    a = an; b = bn;
  }
  return wmma_bf16(a, b, acc);
}

// ---------- kernels ----------

// Pack W[K,Ncols] (row-major fp32) into WMMA B-fragment order, bf16, zero-pad K->Kpad.
__global__ void pignn_pack_w(const float* __restrict__ W, unsigned short* __restrict__ out,
                             int K, int Kpad, int Ncols) {
  int idx = blockIdx.x * blockDim.x + threadIdx.x;
  int ntiles = Ncols >> 4;
  int total = (Kpad >> 5) * ntiles * 512;
  if (idx >= total) return;
  int j = idx & 15;
  int lane = (idx >> 4) & 31;
  int tile = idx >> 9;
  int nt = tile % ntiles, kt = tile / ntiles;
  int k = (kt << 5) + ((lane >> 4) << 3) + (j & 7) + ((j >> 3) << 4);
  int n = (nt << 4) + (lane & 15);
  float v = (k < K) ? W[k * Ncols + n] : 0.0f;
  out[idx] = f2bf(v);
}

__global__ void pignn_zero(float* __restrict__ p, int n) {
  int i = blockIdx.x * blockDim.x + threadIdx.x;
  if (i < n) p[i] = 0.0f;
}

// Encoder: Y[16x128] = relu(X[16xKIN] @ W1 + b1) @ W2 + b2   (KIN zero-padded to 32)
template <int KIN>
__global__ void __launch_bounds__(256, 1)
pignn_enc(const float* __restrict__ X,
          const unsigned short* __restrict__ W1p, const float* __restrict__ b1,
          const unsigned short* __restrict__ W2p, const float* __restrict__ b2,
          float* __restrict__ Y) {
  __shared__ __align__(16) unsigned short A[16 * 40];
  __shared__ __align__(16) unsigned short Hl[16 * 136];
  const int row0 = blockIdx.x << 4;
  const int lane = threadIdx.x & 31, wave = threadIdx.x >> 5;
  const int half = lane >> 4, ncol = lane & 15;
  const int n = (wave << 4) + ncol;

  for (int i = threadIdx.x; i < 16 * 32; i += 256) {
    int r = i >> 5, c = i & 31;
    float v = (c < KIN) ? X[(row0 + r) * KIN + c] : 0.0f;
    A[r * 40 + c] = f2bf(v);
  }
  __syncthreads();

  v8f acc = gemm_lds_gmemw<1>(A, 40, W1p, 8, wave, lane);
  {
    const float bb = b1[n];
#pragma unroll
    for (int v = 0; v < 8; ++v)
      Hl[(v + (half << 3)) * 136 + n] = f2bf(fmaxf(acc[v] + bb, 0.0f));
  }
  __syncthreads();
  v8f acc2 = gemm_lds_gmemw<4>(Hl, 136, W2p, 8, wave, lane);
  const float bb2 = b2[n];
#pragma unroll
  for (int v = 0; v < 8; ++v)
    Y[(size_t)(row0 + v + (half << 3)) * HDIM + n] = acc2[v] + bb2;
}

// Message MLP: per edge tile gather concat(h[src],h[dst],e) -> MLP -> atomic segsum.
// MSG_T tiles per block; only W1 fragments (12 x 8 VGPRs) persistent in registers;
// W2 fragments streamed from L2 per tile.
__global__ void __launch_bounds__(256, 1)
pignn_msg(const float* __restrict__ h, const float* __restrict__ eenc,
          const int* __restrict__ eidx,
          const unsigned short* __restrict__ W1p, const float* __restrict__ b1,
          const unsigned short* __restrict__ W2p, const float* __restrict__ b2,
          float* __restrict__ agg) {
  __shared__ __align__(16) unsigned short A[16 * 392];
  __shared__ __align__(16) unsigned short Hl[16 * 136];
  __shared__ int s_src[16];
  __shared__ int s_dst[2][16];           // double buffer: emit of tile t reads buf t&1
  const int lane = threadIdx.x & 31, wave = threadIdx.x >> 5;
  const int half = lane >> 4, ncol = lane & 15;
  const int n = (wave << 4) + ncol;

  v16bf w1[12];
#pragma unroll
  for (int kt = 0; kt < 12; ++kt) w1[kt] = ld_b_frag(W1p, kt, wave, 8, lane);
  const float bb1 = b1[n], bb2 = b2[n];

  const float4* h4 = (const float4*)h;
  const float4* e4 = (const float4*)eenc;
  const int ntile = NE / 16;
  const int tile0 = blockIdx.x * MSG_T;

  for (int t = 0; t < MSG_T; ++t) {
    const int tile = tile0 + t;
    if (tile >= ntile) break;            // uniform across block
    const int buf = t & 1;
    const int e0 = tile << 4;
    if (threadIdx.x < 16) {
      s_src[threadIdx.x] = eidx[e0 + threadIdx.x];
      s_dst[buf][threadIdx.x] = eidx[NE + e0 + threadIdx.x];
    }
    __syncthreads();                     // B1: indices visible
    for (int i = threadIdx.x; i < 16 * 96; i += 256) {
      int r = i / 96, q = i - r * 96;
      float4 v;
      if (q < 32)      v = h4[(size_t)s_src[r] * 32 + q];
      else if (q < 64) v = h4[(size_t)s_dst[buf][r] * 32 + (q - 32)];
      else             v = e4[(size_t)(e0 + r) * 32 + (q - 64)];
      *(uint2*)&A[r * 392 + q * 4] = f4_to_bf4(v);
    }
    __syncthreads();                     // B2: A tile staged

    v8f acc = gemm_lds_regw<12>(A, 392, w1, lane);
#pragma unroll
    for (int v = 0; v < 8; ++v)
      Hl[(v + (half << 3)) * 136 + n] = f2bf(fmaxf(acc[v] + bb1, 0.0f));
    __syncthreads();                     // B3: hidden staged

    v8f acc2 = gemm_lds_gmemw<4>(Hl, 136, W2p, 8, wave, lane);
#pragma unroll
    for (int v = 0; v < 8; ++v)
      atomicAdd(&agg[(size_t)s_dst[buf][v + (half << 3)] * HDIM + n], acc2[v] + bb2);
  }
}

// Update MLP: concat(h, agg) -> MLP -> residual add into h. Zeroes agg after reading.
// Only W1 fragments persistent; W2 streamed per tile.
__global__ void __launch_bounds__(256, 1)
pignn_upd(float* __restrict__ h, float* __restrict__ agg,
          const unsigned short* __restrict__ W1p, const float* __restrict__ b1,
          const unsigned short* __restrict__ W2p, const float* __restrict__ b2) {
  __shared__ __align__(16) unsigned short A[16 * 264];
  __shared__ __align__(16) unsigned short Hl[16 * 136];
  const int lane = threadIdx.x & 31, wave = threadIdx.x >> 5;
  const int half = lane >> 4, ncol = lane & 15;
  const int n = (wave << 4) + ncol;

  v16bf w1[8];
#pragma unroll
  for (int kt = 0; kt < 8; ++kt) w1[kt] = ld_b_frag(W1p, kt, wave, 8, lane);
  const float bb1 = b1[n], bb2 = b2[n];

  const float4* h4 = (const float4*)h;
  float4* a4 = (float4*)agg;
  const float4 z4 = make_float4(0.f, 0.f, 0.f, 0.f);
  const int ntile = NN / 16;
  const int tile0 = blockIdx.x * UPD_T;

  for (int t = 0; t < UPD_T; ++t) {
    const int tile = tile0 + t;
    if (tile >= ntile) break;            // uniform across block
    const int r0 = tile << 4;
    for (int i = threadIdx.x; i < 16 * 64; i += 256) {
      int r = i >> 6, q = i & 63;
      float4 v;
      if (q < 32) {
        v = h4[(size_t)(r0 + r) * 32 + q];
      } else {
        size_t ix = (size_t)(r0 + r) * 32 + (q - 32);
        v = a4[ix];
        a4[ix] = z4;                     // fold next-layer agg zeroing in here
      }
      *(uint2*)&A[r * 264 + q * 4] = f4_to_bf4(v);
    }
    __syncthreads();                     // B1: A tile staged

    v8f acc = gemm_lds_regw<8>(A, 264, w1, lane);
#pragma unroll
    for (int v = 0; v < 8; ++v)
      Hl[(v + (half << 3)) * 136 + n] = f2bf(fmaxf(acc[v] + bb1, 0.0f));
    __syncthreads();                     // B2: hidden staged

    v8f acc2 = gemm_lds_gmemw<4>(Hl, 136, W2p, 8, wave, lane);
#pragma unroll
    for (int v = 0; v < 8; ++v)
      h[(size_t)(r0 + v + (half << 3)) * HDIM + n] += acc2[v] + bb2;
  }
}

// Decoder: 128 -> 128 (relu, 8 waves WMMA) -> 64 (relu, 4 waves WMMA) -> 3 (+BC masks)
__global__ void __launch_bounds__(256, 1)
pignn_dec(const float* __restrict__ h,
          const unsigned short* __restrict__ W1p, const float* __restrict__ b1,
          const unsigned short* __restrict__ W2p, const float* __restrict__ b2,
          const float* __restrict__ W3, const float* __restrict__ b3,
          const float* __restrict__ bc_disp, const float* __restrict__ bc_rot,
          float* __restrict__ out) {
  __shared__ __align__(16) unsigned short A[16 * 136];
  __shared__ __align__(16) unsigned short Hl[16 * 136];
  __shared__ float D2[16 * 72];
  const int row0 = blockIdx.x << 4;
  const float4* h4 = (const float4*)h;
  for (int i = threadIdx.x; i < 16 * 32; i += 256) {
    int r = i >> 5, q = i & 31;
    *(uint2*)&A[r * 136 + q * 4] = f4_to_bf4(h4[(size_t)(row0 + r) * 32 + q]);
  }
  __syncthreads();
  const int lane = threadIdx.x & 31, wave = threadIdx.x >> 5;
  const int half = lane >> 4, ncol = lane & 15;
  {
    v8f acc = gemm_lds_gmemw<4>(A, 136, W1p, 8, wave, lane);
    int n = (wave << 4) + ncol;
    float bb = b1[n];
#pragma unroll
    for (int v = 0; v < 8; ++v)
      Hl[(v + (half << 3)) * 136 + n] = f2bf(fmaxf(acc[v] + bb, 0.0f));
  }
  __syncthreads();
  if (wave < 4) {                       // 64 output columns
    v8f acc = gemm_lds_gmemw<4>(Hl, 136, W2p, 4, wave, lane);
    int n = (wave << 4) + ncol;
    float bb = b2[n];
#pragma unroll
    for (int v = 0; v < 8; ++v)
      D2[(v + (half << 3)) * 72 + n] = fmaxf(acc[v] + bb, 0.0f);
  }
  __syncthreads();
  if (threadIdx.x < 48) {               // 16 rows x 3 outputs
    int r = threadIdx.x / 3, c = threadIdx.x - r * 3;
    float s = b3[c];
#pragma unroll 8
    for (int k = 0; k < 64; ++k) s += D2[r * 72 + k] * W3[k * 3 + c];
    int node = row0 + r;
    float mask = (c < 2) ? (1.0f - bc_disp[node * 2 + c]) : (1.0f - bc_rot[node]);
    out[(size_t)node * 3 + c] = s * mask;
  }
}

// ---------- host ----------

extern "C" void kernel_launch(void* const* d_in, const int* in_sizes, int n_in,
                              void* d_out, int out_size, void* d_ws, size_t ws_size,
                              hipStream_t stream) {
  (void)in_sizes; (void)n_in; (void)out_size; (void)ws_size;
  const float* x        = (const float*)d_in[0];
  const float* eattr    = (const float*)d_in[1];
  const int*   eidx     = (const int*)d_in[2];
  const float* bc_disp  = (const float*)d_in[3];
  const float* bc_rot   = (const float*)d_in[4];
  const float* enc_n_W1 = (const float*)d_in[5];
  const float* enc_n_b1 = (const float*)d_in[6];
  const float* enc_n_W2 = (const float*)d_in[7];
  const float* enc_n_b2 = (const float*)d_in[8];
  const float* enc_e_W1 = (const float*)d_in[9];
  const float* enc_e_b1 = (const float*)d_in[10];
  const float* enc_e_W2 = (const float*)d_in[11];
  const float* enc_e_b2 = (const float*)d_in[12];
  const float* mW1 = (const float*)d_in[13];
  const float* mb1 = (const float*)d_in[14];
  const float* mW2 = (const float*)d_in[15];
  const float* mb2 = (const float*)d_in[16];
  const float* uW1 = (const float*)d_in[17];
  const float* ub1 = (const float*)d_in[18];
  const float* uW2 = (const float*)d_in[19];
  const float* ub2 = (const float*)d_in[20];
  const float* dW1 = (const float*)d_in[21];
  const float* db1 = (const float*)d_in[22];
  const float* dW2 = (const float*)d_in[23];
  const float* db2 = (const float*)d_in[24];
  const float* dW3 = (const float*)d_in[25];
  const float* db3 = (const float*)d_in[26];

  float* h    = (float*)d_ws;
  float* eenc = h + (size_t)NN * HDIM;
  float* agg  = eenc + (size_t)NE * HDIM;
  unsigned short* pk = (unsigned short*)(agg + (size_t)NN * HDIM);

  const size_t S32     = 1 * 8 * 512;    // Kpad=32,  N=128
  const size_t S128    = 4 * 8 * 512;    // Kpad=128, N=128
  const size_t S384    = 12 * 8 * 512;   // Kpad=384, N=128
  const size_t S256    = 8 * 8 * 512;    // Kpad=256, N=128
  const size_t S128x64 = 4 * 4 * 512;    // Kpad=128, N=64

  unsigned short* p_enW1 = pk; pk += S32;
  unsigned short* p_enW2 = pk; pk += S128;
  unsigned short* p_eeW1 = pk; pk += S32;
  unsigned short* p_eeW2 = pk; pk += S128;
  unsigned short* p_mW1[NLAYERS]; unsigned short* p_mW2[NLAYERS];
  unsigned short* p_uW1[NLAYERS]; unsigned short* p_uW2[NLAYERS];
  for (int l = 0; l < NLAYERS; ++l) {
    p_mW1[l] = pk; pk += S384;
    p_mW2[l] = pk; pk += S128;
    p_uW1[l] = pk; pk += S256;
    p_uW2[l] = pk; pk += S128;
  }
  unsigned short* p_dW1 = pk; pk += S128;
  unsigned short* p_dW2 = pk; pk += S128x64;

  auto pack = [&](const float* W, unsigned short* out, int K, int Kpad, int Ncols) {
    int total = (Kpad / 32) * (Ncols / 16) * 512;
    pignn_pack_w<<<(total + 255) / 256, 256, 0, stream>>>(W, out, K, Kpad, Ncols);
  };

  pack(enc_n_W1, p_enW1, 9, 32, 128);
  pack(enc_n_W2, p_enW2, 128, 128, 128);
  pack(enc_e_W1, p_eeW1, 10, 32, 128);
  pack(enc_e_W2, p_eeW2, 128, 128, 128);
  for (int l = 0; l < NLAYERS; ++l) {
    pack(mW1 + (size_t)l * 384 * 128, p_mW1[l], 384, 384, 128);
    pack(mW2 + (size_t)l * 128 * 128, p_mW2[l], 128, 128, 128);
    pack(uW1 + (size_t)l * 256 * 128, p_uW1[l], 256, 256, 128);
    pack(uW2 + (size_t)l * 128 * 128, p_uW2[l], 128, 128, 128);
  }
  pack(dW1, p_dW1, 128, 128, 128);
  pack(dW2, p_dW2, 128, 128, 64);

  // Initial agg zero (ws is poisoned); subsequent zeroing folded into upd kernel.
  const int aggN = NN * HDIM;
  pignn_zero<<<(aggN + 255) / 256, 256, 0, stream>>>(agg, aggN);

  // Encoders
  pignn_enc<9><<<NN / 16, 256, 0, stream>>>(x, p_enW1, enc_n_b1, p_enW2, enc_n_b2, h);
  pignn_enc<10><<<NE / 16, 256, 0, stream>>>(eattr, p_eeW1, enc_e_b1, p_eeW2, enc_e_b2, eenc);

  // Message-passing layers
  const int msg_blocks = (NE / 16 + MSG_T - 1) / MSG_T;
  const int upd_blocks = (NN / 16 + UPD_T - 1) / UPD_T;
  for (int l = 0; l < NLAYERS; ++l) {
    pignn_msg<<<msg_blocks, 256, 0, stream>>>(h, eenc, eidx,
                                              p_mW1[l], mb1 + l * HDIM,
                                              p_mW2[l], mb2 + l * HDIM, agg);
    pignn_upd<<<upd_blocks, 256, 0, stream>>>(h, agg,
                                              p_uW1[l], ub1 + l * HDIM,
                                              p_uW2[l], ub2 + l * HDIM);
  }

  // Decoder + boundary conditions
  pignn_dec<<<NN / 16, 256, 0, stream>>>(h, p_dW1, db1, p_dW2, db2, dW3, db3,
                                         bc_disp, bc_rot, (float*)d_out);
}
